// LSTM2MLP_17884243820657
// MI455X (gfx1250) — matrix operand order repaired and checked by reading, not compiled, not run on hardware
//
#include <hip/hip_runtime.h>
#include <math.h>

// ---------------------------------------------------------------------------
// MI455X (gfx1250) bidirectional 2-layer LSTM + MLP loss.
// f16 WMMA (v_wmma_f32_16x16x32_f16) for all GEMMs, f32 accumulation,
// f32 pointwise. Recurrence fused (4 gates + nonlinearity) into one kernel
// launch per time step; h_prev ping-pong buffered across launches.
// LDS staging uses CDNA5 async DMA: global_load_async_to_lds_b128 + ASYNCcnt.
// ---------------------------------------------------------------------------

typedef __attribute__((ext_vector_type(16))) _Float16 v16h;
typedef __attribute__((ext_vector_type(8)))  _Float16 v8h;
typedef __attribute__((ext_vector_type(8)))  float    v8f;

// Problem constants
#define BB   64            // batch
#define TT   256           // seq len
#define EE   300           // embed dim
#define EP   320           // embed dim padded to K-multiple of 32
#define HH   512           // hidden
#define GG   2048          // 4*H (gates, PyTorch order i,f,g,o)
#define BT   16384         // B*T
#define LDK  40            // LDS row stride in halfs (32 + 8 pad)

#define USE_ASYNC_LDS 1    // CDNA5 GLOBAL_LOAD_ASYNC_TO_LDS_B128 staging

__device__ __forceinline__ v8f zero8() {
    v8f z;
#pragma unroll
    for (int i = 0; i < 8; ++i) z[i] = 0.0f;
    return z;
}

// 16-byte global -> LDS copy. Async DMA path (ASYNCcnt) with VGPR fallback.
__device__ __forceinline__ void cp16_lds(_Float16* lds_dst, const _Float16* gsrc) {
#if USE_ASYNC_LDS
    uint32_t laddr = (uint32_t)(uintptr_t)lds_dst;   // LDS aperture: offset in [31:0]
    asm volatile("global_load_async_to_lds_b128 %0, %1, off"
                 :: "v"(laddr), "v"(gsrc)
                 : "memory");
#else
    *(v8h*)lds_dst = *(const v8h*)gsrc;
#endif
}

__device__ __forceinline__ void async_wait0() {
#if USE_ASYNC_LDS
    asm volatile("s_wait_asynccnt 0x0" ::: "memory");
#endif
}

// A fragment (16x32 f16, M x K), ISA 7.12.2 layout:
// lane = hi*16 + m16; halfs 0..7 <- K = hi*8 + 0..7 ; halfs 8..15 <- K = 16 + hi*8 + 0..7
__device__ __forceinline__ v16h load_a_frag(const _Float16* base, int ldk, int lane) {
    const int m  = lane & 15;
    const int hi = lane >> 4;
    union { v16h v; v8h h[2]; } u;
    u.h[0] = *(const v8h*)(base + m * ldk + hi * 8);
    u.h[1] = *(const v8h*)(base + m * ldk + 16 + hi * 8);
    return u.v;
}

// B fragment (32x16 f16, K x N) loaded from an N-major tile Bs[n][k]:
// lane = hi*16 + n16; halfs 0..7 <- K = hi*8 + 0..7 ; halfs 8..15 <- K = 16 + hi*8 + 0..7 for column n.
__device__ __forceinline__ v16h load_b_frag(const _Float16* base, int ldk, int lane) {
    const int n  = lane & 15;
    const int hi = lane >> 4;
    union { v16h v; v8h h[2]; } u;
    const _Float16* p = base + n * ldk + hi * 8;
    u.h[0] = *(const v8h*)(p);
    u.h[1] = *(const v8h*)(p + 16);
    return u.v;
}

__device__ __forceinline__ v8f wmma16(v16h a, v16h b, v8f c) {
    return __builtin_amdgcn_wmma_f32_16x16x32_f16(false, a, false, b, (short)0, c,
                                                  false, false);
}

__device__ __forceinline__ float sigmoidf_(float x) { return 1.0f / (1.0f + expf(-x)); }

// ---------------------------------------------------------------------------
// Utility kernels
// ---------------------------------------------------------------------------
__global__ void lstm_zero_kernel(uint32_t* p, int n) {
    int i = blockIdx.x * blockDim.x + threadIdx.x;
    if (i < n) p[i] = 0u;
}

// emb gather: x16[bt][e] = (e < E) ? emb[ids[bt]*E + e] : 0   (f16, padded to EP)
__global__ void lstm_gather_kernel(const int* __restrict__ ids,
                                   const float* __restrict__ emb,
                                   _Float16* __restrict__ x16) {
    int i  = blockIdx.x * 256 + threadIdx.x;       // BT*EP total, exact multiple
    int bt = i / EP;
    int e  = i - bt * EP;
    float v = (e < EE) ? emb[(size_t)ids[bt] * EE + e] : 0.0f;
    x16[i] = (_Float16)v;
}

// weight f32 [N][D] -> f16 [N][Dp] (zero pad)
__global__ void lstm_wcvt_kernel(const float* __restrict__ src,
                                 _Float16* __restrict__ dst,
                                 int N, int D, int Dp) {
    int i = blockIdx.x * 256 + threadIdx.x;
    if (i >= N * Dp) return;
    int n = i / Dp;
    int d = i - n * Dp;
    dst[i] = (d < D) ? (_Float16)src[(size_t)n * D + d] : (_Float16)0.0f;
}

// ---------------------------------------------------------------------------
// Pre-activation GEMM: Out[d][m][n] = x[m][:] . W[d][n][:] + bih[n] + bhh[n]
// A: [M][Kp] f16 (row major), W: [2][N=2048][Kp] f16 (N-major == B layout).
// Block tile 128x128, 8 waves of 64x32, LDS staged (async DMA), K step 32.
// ---------------------------------------------------------------------------
__global__ __launch_bounds__(256) void lstm_gemm_pre_kernel(
    const _Float16* __restrict__ A,
    const _Float16* __restrict__ Wd,
    const float* __restrict__ bih_f, const float* __restrict__ bhh_f,
    const float* __restrict__ bih_b, const float* __restrict__ bhh_b,
    _Float16* __restrict__ Out,
    int M, int Kp) {
    const int dir = blockIdx.z;
    const float* bih = dir ? bih_b : bih_f;
    const float* bhh = dir ? bhh_b : bhh_f;
    const _Float16* W = Wd + (size_t)dir * GG * Kp;
    _Float16* O       = Out + (size_t)dir * M * GG;

    const int n0 = blockIdx.x * 128;
    const int m0 = blockIdx.y * 128;

    __shared__ __align__(16) _Float16 As[128 * LDK];
    __shared__ __align__(16) _Float16 Bs[128 * LDK];

    const int tid  = threadIdx.x;
    const int wave = tid >> 5;
    const int lane = tid & 31;
    const int wm   = wave >> 2;   // 0..1  (64-row group)
    const int wn   = wave & 3;    // 0..3  (32-col group)

    v8f acc[4][2];
#pragma unroll
    for (int i = 0; i < 4; ++i) { acc[i][0] = zero8(); acc[i][1] = zero8(); }

    for (int kb = 0; kb < Kp; kb += 32) {
        __syncthreads();          // LDS reuse: all fragment reads done
        {   // stage A + B: each thread DMAs 2x16B of each tile
            int row = tid >> 1;
            int ko  = (tid & 1) * 16;
            const _Float16* sa = A + (size_t)(m0 + row) * Kp + kb + ko;
            cp16_lds(As + row * LDK + ko,     sa);
            cp16_lds(As + row * LDK + ko + 8, sa + 8);
            const _Float16* sb = W + (size_t)(n0 + row) * Kp + kb + ko;
            cp16_lds(Bs + row * LDK + ko,     sb);
            cp16_lds(Bs + row * LDK + ko + 8, sb + 8);
        }
        async_wait0();            // ASYNCcnt == 0: DMA data landed in LDS
        __syncthreads();
        v16h bf0 = load_b_frag(Bs + (wn * 32) * LDK, LDK, lane);
        v16h bf1 = load_b_frag(Bs + (wn * 32 + 16) * LDK, LDK, lane);
#pragma unroll
        for (int ms = 0; ms < 4; ++ms) {
            v16h af = load_a_frag(As + (wm * 64 + ms * 16) * LDK, LDK, lane);
            acc[ms][0] = wmma16(af, bf0, acc[ms][0]);
            acc[ms][1] = wmma16(af, bf1, acc[ms][1]);
        }
    }

    const int hi = lane >> 4, nl = lane & 15;
#pragma unroll
    for (int ms = 0; ms < 4; ++ms)
#pragma unroll
        for (int ns = 0; ns < 2; ++ns) {
            int n = n0 + wn * 32 + ns * 16 + nl;
            float bsum = bih[n] + bhh[n];
#pragma unroll
            for (int r = 0; r < 8; ++r) {
                int m = m0 + wm * 64 + ms * 16 + r + hi * 8;
                O[(size_t)m * GG + n] = (_Float16)(acc[ms][ns][r] + bsum);
            }
        }
}

// ---------------------------------------------------------------------------
// Fused LSTM step: gates = h_prev @ Whh^T + pre[:,t,:]; c,h update.
// Grid: (16 col-blocks of 32 h-cols, 2 directions). Block 256 thr / 8 waves.
// Block GEMM tile: M=64(all batch) x N=128 (4 gates x 32 cols), K=512.
// h_prev ping-pong: read hin, write hout (separate launches serialize steps).
// ---------------------------------------------------------------------------
__global__ __launch_bounds__(256) void lstm_step_kernel(
    const _Float16* __restrict__ whh,    // [2][2048][512] f16 (this layer)
    const _Float16* __restrict__ pre,    // [2][BT][2048] f16
    const _Float16* __restrict__ hin,    // [2][64][512] f16
    _Float16* __restrict__ hout,         // [2][64][512] f16
    float* __restrict__ cst,             // [2][64][512] f32
    _Float16* __restrict__ x1out,        // layer0: [BT][1024] f16, else null
    float* __restrict__ hsout,           // layer1: [2][BT][512] f32, else null
    int s) {
    const int dir = blockIdx.y;
    const int t   = dir ? (TT - 1 - s) : s;
    const int cb  = blockIdx.x;          // 32 h-cols per block

    __shared__ __align__(16) _Float16 As[64 * LDK];
    __shared__ __align__(16) _Float16 Bs[128 * LDK];
    __shared__ float gbuf[4][64][33];

    const int tid  = threadIdx.x;
    const int wave = tid >> 5;
    const int lane = tid & 31;

    const _Float16* hp = hin + (size_t)dir * BB * HH;
    const _Float16* Wl = whh + (size_t)dir * GG * HH;

    v8f acc[4];
#pragma unroll
    for (int i = 0; i < 4; ++i) acc[i] = zero8();

    for (int kb = 0; kb < HH; kb += 32) {
        __syncthreads();
        {   // A: 64 rows x 32 halfs (one 16B DMA/thread)
            int ra = tid >> 2, ka = (tid & 3) * 8;
            cp16_lds(As + ra * LDK + ka, hp + ra * HH + kb + ka);
            // B: 128 gate-rows x 32 halfs (two 16B DMAs/thread)
            int n  = tid >> 1, kbo = (tid & 1) * 16;
            int grow = (n >> 5) * HH + cb * 32 + (n & 31); // gate*512 + hcol
            const _Float16* sb = Wl + (size_t)grow * HH + kb + kbo;
            cp16_lds(Bs + n * LDK + kbo,     sb);
            cp16_lds(Bs + n * LDK + kbo + 8, sb + 8);
        }
        async_wait0();
        __syncthreads();
        v16h bf = load_b_frag(Bs + (wave * 16) * LDK, LDK, lane);
#pragma unroll
        for (int ms = 0; ms < 4; ++ms) {
            v16h af = load_a_frag(As + ms * 16 * LDK, LDK, lane);
            acc[ms] = wmma16(af, bf, acc[ms]);
        }
    }

    // scatter accumulators to gate buffer: wave -> (gate = wave>>1, 16 cols)
    {
        const int g   = wave >> 1;
        const int sub = wave & 1;
        const int hi  = lane >> 4;
        const int nl  = lane & 15;
        const int nc  = sub * 16 + nl;
#pragma unroll
        for (int ms = 0; ms < 4; ++ms)
#pragma unroll
            for (int r = 0; r < 8; ++r)
                gbuf[g][ms * 16 + r + hi * 8][nc] = acc[ms][r];
    }
    __syncthreads();

    // pointwise: 64 x 32 outputs, 8 per thread
#pragma unroll
    for (int j = 0; j < 8; ++j) {
        int o   = j * 256 + tid;
        int m   = o >> 5;
        int col = o & 31;
        int h   = cb * 32 + col;
        size_t bt    = (size_t)m * TT + t;
        size_t pbase = ((size_t)dir * BT + bt) * GG + h;
        float iv = gbuf[0][m][col] + (float)pre[pbase];
        float fv = gbuf[1][m][col] + (float)pre[pbase + 512];
        float gv = gbuf[2][m][col] + (float)pre[pbase + 1024];
        float ov = gbuf[3][m][col] + (float)pre[pbase + 1536];
        size_t ci = (size_t)dir * BB * HH + (size_t)m * HH + h;
        float cold = cst[ci];
        float cnew = sigmoidf_(fv) * cold + sigmoidf_(iv) * tanhf(gv);
        float hv   = sigmoidf_(ov) * tanhf(cnew);
        cst[ci]  = cnew;
        hout[ci] = (_Float16)hv;
        if (x1out) x1out[bt * (2 * HH) + (size_t)dir * HH + h] = (_Float16)hv;
        if (hsout) hsout[((size_t)dir * BT + bt) * HH + h] = hv;
    }
}

// ---------------------------------------------------------------------------
// Score: one wave per (b,t): err[bt] = (sigmoid(0.5*(hf+hb).w + b) - label)^2 * mask
// ---------------------------------------------------------------------------
__global__ __launch_bounds__(256) void lstm_score_kernel(
    const float* __restrict__ hs1, const float* __restrict__ lin_w,
    const float* __restrict__ lin_b, const float* __restrict__ labels,
    const float* __restrict__ masks, float* __restrict__ err) {
    int gid  = blockIdx.x * 8 + (threadIdx.x >> 5);   // bt index
    int lane = threadIdx.x & 31;
    const float* hf = hs1 + (size_t)gid * HH;
    const float* hb = hs1 + (size_t)BT * HH + (size_t)gid * HH;
    float sum = 0.0f;
    for (int h = lane; h < HH; h += 32)
        sum += 0.5f * (hf[h] + hb[h]) * lin_w[h];
#pragma unroll
    for (int off = 16; off > 0; off >>= 1)
        sum += __shfl_down(sum, off, 32);
    if (lane == 0) {
        float sc = sigmoidf_(sum + lin_b[0]);
        float d  = sc - labels[gid];
        err[gid] = d * d * masks[gid];
    }
}

__global__ void lstm_loss_kernel(const float* __restrict__ err,
                                 const float* __restrict__ masks,
                                 float* __restrict__ out) {
    __shared__ float part[BB];
    int b = threadIdx.x;   // launched with 64 threads
    float se = 0.0f, sm = 0.0f;
    for (int t = 0; t < TT; ++t) {
        se += err[b * TT + t];
        sm += masks[b * TT + t];
    }
    part[b] = se / sm;
    __syncthreads();
    if (b == 0) {
        float l = 0.0f;
        for (int i = 0; i < BB; ++i) l += part[i];
        out[0] = l / (float)BB;
    }
}

// ---------------------------------------------------------------------------
// Host-side orchestration
// ---------------------------------------------------------------------------
extern "C" void kernel_launch(void* const* d_in, const int* in_sizes, int n_in,
                              void* d_out, int out_size, void* d_ws, size_t ws_size,
                              hipStream_t stream) {
    const int*   ids    = (const int*)d_in[0];
    const float* masks  = (const float*)d_in[1];
    const float* labels = (const float*)d_in[2];
    const float* emb    = (const float*)d_in[3];
    const float* lin_w  = (const float*)d_in[4];
    const float* lin_b  = (const float*)d_in[5];
    // 6..21: (w_ih, w_hh, b_ih, b_hh) for l0f, l0b, l1f, l1b

    char* ws = (char*)d_ws;
    size_t off = 0;
    auto alloc = [&](size_t bytes) -> char* {
        char* p = ws + off;
        off = (off + bytes + 255) & ~(size_t)255;
        return p;
    };

    _Float16* x16    = (_Float16*)alloc((size_t)BT * EP * 2);          // 10.5 MB
    _Float16* wih0   = (_Float16*)alloc((size_t)2 * GG * EP * 2);      // 2.6 MB
    _Float16* wih1   = (_Float16*)alloc((size_t)2 * GG * 1024 * 2);    // 8.4 MB
    _Float16* whh16  = (_Float16*)alloc((size_t)4 * GG * HH * 2);      // 8.4 MB
    _Float16* pre16  = (_Float16*)alloc((size_t)2 * BT * GG * 2);      // 134 MB
    _Float16* x1_16  = (_Float16*)alloc((size_t)BT * 2 * HH * 2);      // 33.6 MB
    float*    hs1    = (float*)alloc((size_t)2 * BT * HH * 4);         // 67 MB
    _Float16* hping  = (_Float16*)alloc((size_t)2 * 2 * BB * HH * 2);  // 2 slots
    float*    cstate = (float*)alloc((size_t)2 * BB * HH * 4);
    float*    errarr = (float*)alloc((size_t)BT * 4);
    (void)ws_size; (void)in_sizes; (void)n_in; (void)out_size;

    const size_t HSLOT = (size_t)2 * BB * HH;   // halfs per ping-pong slot

    // 1) embedding gather -> f16 padded
    lstm_gather_kernel<<<(BT * EP) / 256, 256, 0, stream>>>(ids, emb, x16);

    // 2) weight conversion (keep [N][K] row-major == WMMA B layout)
    lstm_wcvt_kernel<<<(GG * EP + 255) / 256, 256, 0, stream>>>(
        (const float*)d_in[6], wih0, GG, EE, EP);
    lstm_wcvt_kernel<<<(GG * EP + 255) / 256, 256, 0, stream>>>(
        (const float*)d_in[10], wih0 + (size_t)GG * EP, GG, EE, EP);
    lstm_wcvt_kernel<<<(GG * 1024 + 255) / 256, 256, 0, stream>>>(
        (const float*)d_in[14], wih1, GG, 1024, 1024);
    lstm_wcvt_kernel<<<(GG * 1024 + 255) / 256, 256, 0, stream>>>(
        (const float*)d_in[18], wih1 + (size_t)GG * 1024, GG, 1024, 1024);
    lstm_wcvt_kernel<<<(GG * HH + 255) / 256, 256, 0, stream>>>(
        (const float*)d_in[7], whh16 + (size_t)0 * GG * HH, GG, HH, HH);
    lstm_wcvt_kernel<<<(GG * HH + 255) / 256, 256, 0, stream>>>(
        (const float*)d_in[11], whh16 + (size_t)1 * GG * HH, GG, HH, HH);
    lstm_wcvt_kernel<<<(GG * HH + 255) / 256, 256, 0, stream>>>(
        (const float*)d_in[15], whh16 + (size_t)2 * GG * HH, GG, HH, HH);
    lstm_wcvt_kernel<<<(GG * HH + 255) / 256, 256, 0, stream>>>(
        (const float*)d_in[19], whh16 + (size_t)3 * GG * HH, GG, HH, HH);

    // 3) layer-0 pre-activations (both dirs), bias fused
    {
        dim3 grid(GG / 128, BT / 128, 2);
        lstm_gemm_pre_kernel<<<grid, 256, 0, stream>>>(
            x16, wih0, (const float*)d_in[8], (const float*)d_in[9],
            (const float*)d_in[12], (const float*)d_in[13], pre16, BT, EP);
    }

    // 4) layer-0 recurrence (f and b concurrently via grid.y)
    {
        int zn = (int)((2 * HSLOT * 2 + (size_t)2 * BB * HH * 4) / 4);
        lstm_zero_kernel<<<(zn + 255) / 256, 256, 0, stream>>>((uint32_t*)hping, zn);
        dim3 grid(HH / 32, 2);
        for (int s = 0; s < TT; ++s) {
            const _Float16* hin = hping + (size_t)(s & 1) * HSLOT;
            _Float16* hout      = hping + (size_t)((s & 1) ^ 1) * HSLOT;
            lstm_step_kernel<<<grid, 256, 0, stream>>>(
                whh16, pre16, hin, hout, cstate, x1_16, (float*)nullptr, s);
        }
    }

    // 5) layer-1 pre-activations from x1 (f16 concat of h0f|h0b)
    {
        dim3 grid(GG / 128, BT / 128, 2);
        lstm_gemm_pre_kernel<<<grid, 256, 0, stream>>>(
            x1_16, wih1, (const float*)d_in[16], (const float*)d_in[17],
            (const float*)d_in[20], (const float*)d_in[21], pre16, BT, 1024);
    }

    // 6) layer-1 recurrence
    {
        int zn = (int)((2 * HSLOT * 2 + (size_t)2 * BB * HH * 4) / 4);
        lstm_zero_kernel<<<(zn + 255) / 256, 256, 0, stream>>>((uint32_t*)hping, zn);
        dim3 grid(HH / 32, 2);
        for (int s = 0; s < TT; ++s) {
            const _Float16* hin = hping + (size_t)(s & 1) * HSLOT;
            _Float16* hout      = hping + (size_t)((s & 1) ^ 1) * HSLOT;
            lstm_step_kernel<<<grid, 256, 0, stream>>>(
                whh16 + (size_t)2 * GG * HH, pre16, hin, hout, cstate,
                (_Float16*)nullptr, hs1, s);
        }
    }

    // 7) scores + masked MSE
    lstm_score_kernel<<<BT / 8, 256, 0, stream>>>(hs1, lin_w, lin_b, labels,
                                                  masks, errarr);
    lstm_loss_kernel<<<1, BB, 0, stream>>>(errarr, masks, (float*)d_out);
}
